// LocalMultiHeadAttention_90761248899250
// MI455X (gfx1250) — matrix operand order repaired and checked by reading, not compiled, not run on hardware
//
#include <hip/hip_runtime.h>
#include <hip/hip_bf16.h>
#include <math.h>

// ---------------- problem constants ----------------
#define BATCH   4
#define SEQ     1536
#define DMODEL  1024
#define NH      16
#define DH      64
#define WIN     128
#define SPAD    (SEQ + 2*WIN)     // 1792 padded key length
#define NCH     (SEQ / WIN)       // 12 chunks
#define MROWS   (BATCH * SEQ)     // 6144
#define QKW     (2*WIN + 1)       // 257 banded logits
#define SSTR    392               // f32 LDS row stride for 384-wide score strip
#define BSTR    40                // f16 LDS row stride for 32-wide weight slab (bank pad)

typedef _Float16 half_t;
typedef __attribute__((ext_vector_type(8)))  _Float16 v8h;
typedef __attribute__((ext_vector_type(16))) _Float16 v16h;
typedef __attribute__((ext_vector_type(8)))  float    v8f;
typedef __attribute__((ext_vector_type(4)))  int      v4i;

union ABFrag { v16h v; v8h h[2]; };

// ---- gfx1250 async global->LDS staging (ASYNCcnt path), guarded ----
#if __has_builtin(__builtin_amdgcn_global_load_async_to_lds_b128) && \
    __has_builtin(__builtin_amdgcn_s_wait_asynccnt)
#define USE_ASYNC_LDS 1
#else
#define USE_ASYNC_LDS 0
#endif

__device__ inline void stage_b128(const half_t* gsrc, half_t* ldst) {
#if USE_ASYNC_LDS
    __builtin_amdgcn_global_load_async_to_lds_b128(
        (__attribute__((address_space(1))) v4i*)gsrc,
        (__attribute__((address_space(3))) v4i*)ldst, 0, 0);
#else
    *(v8h*)ldst = *(const v8h*)gsrc;   // sync fallback: global_load + ds_store
#endif
}
__device__ inline void async_wait_all() {
#if USE_ASYNC_LDS
    __builtin_amdgcn_s_wait_asynccnt(0);
#endif
}

// A fragment: 16x32 f16 tile, row-major source (lda in halves).
// wave32: m = lane&15, hi = lane>>4; elems 0..7 <- [k0+hi*8], elems 8..15 <- [k0+16+hi*8]
__device__ inline v16h load_a16(const half_t* A, int lda, int m0, int k0, int lane) {
    const int m  = m0 + (lane & 15);
    const int hi = (lane >> 4) & 1;
    const half_t* p = A + (size_t)m * lda + k0 + hi * 8;
    ABFrag u;
    u.h[0] = *(const v8h*)(p);
    u.h[1] = *(const v8h*)(p + 16);
    return u.v;
}

// B fragment: 32x16 f16 tile; source is B-transposed row-major (Bt[n][k], ldb in halves).
// lane: n = lane&15; elems 0..7 <- [k0+hi*8], elems 8..15 <- [k0+16+hi*8]
__device__ inline v16h load_b16(const half_t* Bt, int ldb, int n0, int k0, int lane) {
    const int n  = n0 + (lane & 15);
    const int hi = (lane >> 4) & 1;
    const half_t* p = Bt + (size_t)n * ldb + k0 + hi * 8;
    ABFrag u;
    u.h[0] = *(const v8h*)(p);
    u.h[1] = *(const v8h*)(p + 16);
    return u.v;
}

__device__ inline v8f wmma16(v16h a, v16h b, v8f c) {
    return __builtin_amdgcn_wmma_f32_16x16x32_f16(
        false, a, false, b, (short)0, c, false, false);
}

__device__ inline v8f zero8() {
    v8f z;
#pragma unroll
    for (int e = 0; e < 8; ++e) z[e] = 0.0f;
    return z;
}

// ---------------- utility kernels ----------------
__global__ void fill_zero_u4(uint4* p, int n4) {
    int i = blockIdx.x * blockDim.x + threadIdx.x;
    int stride = gridDim.x * blockDim.x;
    uint4 z = {0u, 0u, 0u, 0u};
    for (; i < n4; i += stride) p[i] = z;
}

__global__ void f32_to_f16_kernel(const float* __restrict__ s, half_t* __restrict__ d, int n) {
    int i = blockIdx.x * blockDim.x + threadIdx.x;
    int stride = gridDim.x * blockDim.x;
    for (; i < n; i += stride) d[i] = (half_t)s[i];
}

// ---------------- fused QKV projection ----------------
// grid: (48 row-blocks of 128, 16 heads), block 256 (8 waves).
// Weight K-slabs (3 matrices x 64 rows x 32 halves) staged in LDS, double-buffered,
// via async global->LDS; each thread stages exactly one b128 chunk per matrix.
__global__ __launch_bounds__(256)
void qkv_proj_kernel(const half_t* __restrict__ xh,
                     const half_t* __restrict__ wq,
                     const half_t* __restrict__ wk,
                     const half_t* __restrict__ wv,
                     const float*  __restrict__ bq,
                     const float*  __restrict__ bv,
                     half_t* __restrict__ qh,
                     half_t* __restrict__ kh,
                     half_t* __restrict__ vt)
{
    __shared__ half_t wslab[2][192 * BSTR];     // 30 KB

    const int rb   = blockIdx.x;
    const int h    = blockIdx.y;
    const int tid  = threadIdx.x;
    const int wid  = tid >> 5;
    const int lane = tid & 31;
    const int row0 = rb * 128 + wid * 16;
    const int hi   = lane >> 4;
    const int nlo  = lane & 15;

    const int srow = (tid >> 2) & 63;           // weight row this thread stages (0..63)
    const int part = tid & 3;                   // which b128 chunk of the 32-half row

    v8f accq[4], acck[4], accv[4];
#pragma unroll
    for (int i = 0; i < 4; ++i) { accq[i] = zero8(); acck[i] = zero8(); accv[i] = zero8(); }

    // stage slab for k0 into wslab[buf]
    auto stage = [&](int k0, int buf) {
        const size_t go = (size_t)(h * DH + srow) * DMODEL + k0 + part * 8;
        stage_b128(wq + go, &wslab[buf][(0 * 64 + srow) * BSTR + part * 8]);
        stage_b128(wk + go, &wslab[buf][(1 * 64 + srow) * BSTR + part * 8]);
        stage_b128(wv + go, &wslab[buf][(2 * 64 + srow) * BSTR + part * 8]);
    };

    stage(0, 0);
    v16h a_cur = load_a16(xh, DMODEL, row0, 0, lane);
    async_wait_all();
    __syncthreads();

    int buf = 0;
    for (int k0 = 0; k0 < DMODEL; k0 += 32) {
        const bool last = (k0 + 32 >= DMODEL);
        if (!last) stage(k0 + 32, buf ^ 1);                 // overlap DMA with math
        v16h a_next = a_cur;
        if (!last) a_next = load_a16(xh, DMODEL, row0, k0 + 32, lane);

        const half_t* bs = &wslab[buf][0];
#pragma unroll
        for (int nt = 0; nt < 4; ++nt) {
            v16h b;
            b = load_b16(bs, BSTR, 0 * 64 + nt * 16, 0, lane); accq[nt] = wmma16(a_cur, b, accq[nt]);
            b = load_b16(bs, BSTR, 1 * 64 + nt * 16, 0, lane); acck[nt] = wmma16(a_cur, b, acck[nt]);
            b = load_b16(bs, BSTR, 2 * 64 + nt * 16, 0, lane); accv[nt] = wmma16(a_cur, b, accv[nt]);
        }
        async_wait_all();
        __syncthreads();
        a_cur = a_next;
        buf ^= 1;
    }

    const float scale = 0.35355339059327373f;   // 64^(-1/4)
#pragma unroll
    for (int nt = 0; nt < 4; ++nt) {
        const int n   = nt * 16 + nlo;          // dh within head
        const float bqv = bq[h * DH + n];
        const float bvv = bv[h * DH + n];
#pragma unroll
        for (int r = 0; r < 8; ++r) {
            const int m  = row0 + r + hi * 8;
            const int bb = m / SEQ;
            const int s  = m % SEQ;
            const int hb = bb * NH + h;
            qh[((size_t)hb * SEQ  + s)       * DH + n]   = (half_t)((accq[nt][r] + bqv) * scale);
            kh[((size_t)hb * SPAD + s + WIN) * DH + n]   = (half_t)(acck[nt][r] * scale);
            vt[((size_t)hb * DH   + n) * SPAD + s + WIN] = (half_t)(accv[nt][r] + bvv);
        }
    }
}

// ---------------- local banded attention ----------------
// grid: BATCH*NH*NCH*2 blocks, 128 threads (4 waves). Each wave owns 16 queries.
__global__ __launch_bounds__(128)
void local_attn_kernel(const half_t* __restrict__ qh,
                       const half_t* __restrict__ kh,
                       const half_t* __restrict__ vt,
                       half_t* __restrict__ ctxh,
                       float*  __restrict__ qk)
{
    extern __shared__ float sbuf[];             // [4 waves][16 rows][SSTR]

    const int t       = blockIdx.x;
    const int half_id = t & 1;
    const int c       = (t >> 1) % NCH;
    const int bh      = t / (2 * NCH);
    const int h       = bh % NH;
    const int b       = bh / NH;
    const int wid     = threadIdx.x >> 5;
    const int lane    = threadIdx.x & 31;
    const int off     = half_id * 64 + wid * 16;
    const int hi      = lane >> 4;
    const int nlo     = lane & 15;

    const half_t* qbase = qh + (size_t)(b * NH + h) * SEQ  * DH;
    const half_t* kbase = kh + (size_t)(b * NH + h) * SPAD * DH;
    const half_t* vbase = vt + (size_t)(b * NH + h) * DH   * SPAD;
    float* srow = sbuf + (size_t)(wid * 16) * SSTR;

    // ---- phase 1: scores 16x384 ----
    v16h a0 = load_a16(qbase, DH, c * 128 + off, 0,  lane);
    v16h a1 = load_a16(qbase, DH, c * 128 + off, 32, lane);
    for (int nt = 0; nt < 24; ++nt) {
        v8f acc = zero8();
        v16h b0 = load_b16(kbase, DH, c * 128 + nt * 16, 0,  lane);
        acc = wmma16(a0, b0, acc);
        v16h b1 = load_b16(kbase, DH, c * 128 + nt * 16, 32, lane);
        acc = wmma16(a1, b1, acc);
#pragma unroll
        for (int r = 0; r < 8; ++r)
            srow[(size_t)(r + hi * 8) * SSTR + nt * 16 + nlo] = acc[r];
    }

    // ---- phase 2: band mask, qk logits, softmax, f16 probs in place ----
    for (int r = 0; r < 16; ++r) {
        const int x = off + r;
        float* row = srow + (size_t)r * SSTR;
        float vals[12];
        float mx = -INFINITY;
#pragma unroll
        for (int tt = 0; tt < 12; ++tt) {
            const int y = lane + tt * 32;
            const float sv = row[y];
            const int j = y - x;
            const int key_pos = c * WIN + y - WIN;
            const bool valid  = (key_pos >= 0) && (key_pos < SEQ);
            const bool inband = (j >= 0) && (j <= 2 * WIN);
            const float masked = (inband && valid) ? sv : -INFINITY;
            if (inband) {
                const int s = c * WIN + x;
                qk[((size_t)(b * SEQ + s) * NH + h) * QKW + j] = valid ? sv : -INFINITY;
            }
            vals[tt] = masked;
            mx = fmaxf(mx, masked);
        }
        for (int o = 16; o >= 1; o >>= 1) mx = fmaxf(mx, __shfl_xor(mx, o, 32));
        float sm = 0.0f;
#pragma unroll
        for (int tt = 0; tt < 12; ++tt) { float e = __expf(vals[tt] - mx); vals[tt] = e; sm += e; }
        for (int o = 16; o >= 1; o >>= 1) sm += __shfl_xor(sm, o, 32);
        const float inv = 1.0f / sm;
        half_t* prow = (half_t*)row;            // in-place f16 (wave-private strip)
#pragma unroll
        for (int tt = 0; tt < 12; ++tt)
            prow[lane + tt * 32] = (half_t)(vals[tt] * inv);
    }

    // ---- phase 3: ctx 16x64 = P(16x384) @ V(384x64) ----
    const half_t* pbase = (const half_t*)srow;  // lda = 2*SSTR halves
    v8f acc[4];
#pragma unroll
    for (int i = 0; i < 4; ++i) acc[i] = zero8();
    for (int kk = 0; kk < 384; kk += 32) {
        v16h a = load_a16(pbase, 2 * SSTR, 0, kk, lane);
#pragma unroll
        for (int nt = 0; nt < 4; ++nt) {
            v16h bb = load_b16(vbase, SPAD, nt * 16, c * 128 + kk, lane);
            acc[nt] = wmma16(a, bb, acc[nt]);
        }
    }
#pragma unroll
    for (int nt = 0; nt < 4; ++nt) {
        const int dh = nt * 16 + nlo;
#pragma unroll
        for (int r = 0; r < 8; ++r) {
            const int s = c * 128 + off + r + hi * 8;
            ctxh[(size_t)(b * SEQ + s) * DMODEL + h * DH + dh] = (half_t)acc[nt][r];
        }
    }
}

// ---------------- output projection ----------------
// grid (48 row-blocks, 8 col-blocks of 128), block 256 (8 waves).
// Wo K-slab (128 rows x 32 halves) staged in LDS, double-buffered, async.
__global__ __launch_bounds__(256)
void out_proj_kernel(const half_t* __restrict__ ah,
                     const half_t* __restrict__ wo,
                     const float*  __restrict__ bo,
                     float* __restrict__ out)
{
    __shared__ half_t wslab[2][128 * BSTR];     // 20 KB

    const int rb   = blockIdx.x;
    const int cb   = blockIdx.y;
    const int tid  = threadIdx.x;
    const int wid  = tid >> 5;
    const int lane = tid & 31;
    const int row0 = rb * 128 + wid * 16;
    const int hi   = lane >> 4;
    const int nlo  = lane & 15;

    const int part = tid & 3;

    auto stage = [&](int k0, int buf) {
#pragma unroll
        for (int i = 0; i < 2; ++i) {
            const int row = (tid >> 2) + 64 * i;        // 0..127
            stage_b128(wo + (size_t)(cb * 128 + row) * DMODEL + k0 + part * 8,
                       &wslab[buf][row * BSTR + part * 8]);
        }
    };

    v8f acc[8];
#pragma unroll
    for (int i = 0; i < 8; ++i) acc[i] = zero8();

    stage(0, 0);
    v16h a_cur = load_a16(ah, DMODEL, row0, 0, lane);
    async_wait_all();
    __syncthreads();

    int buf = 0;
    for (int k0 = 0; k0 < DMODEL; k0 += 32) {
        const bool last = (k0 + 32 >= DMODEL);
        if (!last) stage(k0 + 32, buf ^ 1);
        v16h a_next = a_cur;
        if (!last) a_next = load_a16(ah, DMODEL, row0, k0 + 32, lane);

        const half_t* bs = &wslab[buf][0];
#pragma unroll
        for (int j = 0; j < 8; ++j) {
            v16h b = load_b16(bs, BSTR, j * 16, 0, lane);
            acc[j] = wmma16(a_cur, b, acc[j]);
        }
        async_wait_all();
        __syncthreads();
        a_cur = a_next;
        buf ^= 1;
    }

#pragma unroll
    for (int j = 0; j < 8; ++j) {
        const int col = cb * 128 + j * 16 + nlo;
        const float bv = bo[col];
#pragma unroll
        for (int r = 0; r < 8; ++r) {
            const int m = row0 + r + hi * 8;
            out[(size_t)m * DMODEL + col] = acc[j][r] + bv;
        }
    }
}

// ---------------- host launcher ----------------
extern "C" void kernel_launch(void* const* d_in, const int* in_sizes, int n_in,
                              void* d_out, int out_size, void* d_ws, size_t ws_size,
                              hipStream_t stream) {
    const float* x  = (const float*)d_in[0];
    const float* Wq = (const float*)d_in[1];
    const float* bq = (const float*)d_in[2];
    const float* Wk = (const float*)d_in[3];
    const float* Wv = (const float*)d_in[4];
    const float* bv = (const float*)d_in[5];
    const float* Wo = (const float*)d_in[6];
    const float* bo = (const float*)d_in[7];

    float* out = (float*)d_out;
    float* qk  = out + (size_t)MROWS * DMODEL;

    half_t* ws = (half_t*)d_ws;
    size_t off = 0;
    half_t* XH  = ws + off; off += (size_t)MROWS * DMODEL;
    half_t* WQH = ws + off; off += (size_t)DMODEL * DMODEL;
    half_t* WKH = ws + off; off += (size_t)DMODEL * DMODEL;
    half_t* WVH = ws + off; off += (size_t)DMODEL * DMODEL;
    half_t* WOH = ws + off; off += (size_t)DMODEL * DMODEL;
    half_t* QH  = ws + off; off += (size_t)BATCH * NH * SEQ  * DH;
    half_t* KH  = ws + off; off += (size_t)BATCH * NH * SPAD * DH;
    half_t* VT  = ws + off; off += (size_t)BATCH * NH * DH * SPAD;
    half_t* CTX = ws + off; off += (size_t)MROWS * DMODEL;

    // 1) zero padded K and Vt buffers (contiguous region)
    const size_t padHalves = 2ull * BATCH * NH * SPAD * DH;
    fill_zero_u4<<<1024, 256, 0, stream>>>((uint4*)KH, (int)(padHalves * sizeof(half_t) / 16));

    // 2) f32 -> f16 casts
    f32_to_f16_kernel<<<2048, 256, 0, stream>>>(x,  XH,  MROWS * DMODEL);
    f32_to_f16_kernel<<<1024, 256, 0, stream>>>(Wq, WQH, DMODEL * DMODEL);
    f32_to_f16_kernel<<<1024, 256, 0, stream>>>(Wk, WKH, DMODEL * DMODEL);
    f32_to_f16_kernel<<<1024, 256, 0, stream>>>(Wv, WVH, DMODEL * DMODEL);
    f32_to_f16_kernel<<<1024, 256, 0, stream>>>(Wo, WOH, DMODEL * DMODEL);

    // 3) fused QKV projection
    qkv_proj_kernel<<<dim3(MROWS / 128, NH), 256, 0, stream>>>(
        XH, WQH, WKH, WVH, bq, bv, QH, KH, VT);

    // 4) local banded attention (+ qk logits)
    const size_t ldsBytes = 4ull * 16 * SSTR * sizeof(float);   // ~100 KB
    local_attn_kernel<<<dim3(BATCH * NH * NCH * 2), 128, ldsBytes, stream>>>(
        QH, KH, VT, CTX, qk);

    // 5) output projection with bias
    out_proj_kernel<<<dim3(MROWS / 128, DMODEL / 128), 256, 0, stream>>>(
        CTX, WOH, bo, out);
}
